// Cross_Transformer_29240137351904
// MI455X (gfx1250) — compile-verified, matrix operand order checked by reading, not run on hardware
//
#include <hip/hip_runtime.h>
#include <hip/hip_bf16.h>
#include <math.h>

// ---------------------------------------------------------------------------
// Cross-transformer forward for MI455X (gfx1250, wave32, WMMA).
// GEMMs: v_wmma_f32_16x16x32_bf16, 128x64 block tile, 32x32 register tile
// per wave (4 accumulators, 2x operand reuse), KC=64, padded LDS staging,
// global_prefetch of the next K-chunk. Attention is flash-style (online
// softmax). Epilogues (bias/residual/GELU/LayerNorm) in f32.
// ---------------------------------------------------------------------------

typedef __attribute__((ext_vector_type(16))) __bf16 v16bf;
typedef __attribute__((ext_vector_type(8)))  float  v8f;

#define D_MODEL 768
#define SEQ     1024
#define BATCH   2
#define NHEAD   12
#define DHEAD   64

#define GF_GELU  1
#define GF_ACCUM 2

// K-index pattern for the 16-bit A/B operand layout (16x32 bf16, wave32):
// lanes 0-15: elem i -> K = (i<8 ? i : i+8); lanes 16-31: +8.
__device__ __forceinline__ int kpat(int i, int hi8) {
  return ((i < 8) ? i : (i + 8)) + hi8;
}

__device__ __forceinline__ v8f wmma_bf16(v16bf a, v16bf b, v8f c) {
  return __builtin_amdgcn_wmma_f32_16x16x32_bf16(
      /*neg_a=*/false, a, /*neg_b=*/false, b,
      /*c_mod=*/(short)0, c, /*reuse_a=*/false, /*reuse_b=*/false);
}

// ---------------------------------------------------------------------------
// Generic GEMM: C[M,N] = act( A[M,K](lda) @ W[N,K](ldw)^T + bias + R )
// Block = 256 threads = 8 waves (4 M x 2 N) -> 128x64 block tile; each wave
// computes a 32x32 register tile (4 WMMA accumulators).
// Requires M%128==0, N%64==0, K%64==0 (true for all shapes in this model).
// ---------------------------------------------------------------------------
#define KC   64
#define LDP  72   // padded LDS row: 64 bf16 data + 8 pad (144B, 16B aligned)

__global__ __launch_bounds__(256)
void gemm_wmma_kernel(const float* __restrict__ A, int lda,
                      const float* __restrict__ W, int ldw,
                      const float* __restrict__ bias,
                      const float* __restrict__ R,
                      float* __restrict__ C,
                      int M, int N, int K, int flags)
{
  __shared__ __bf16 As[128][LDP];
  __shared__ __bf16 Ws[64][LDP];

  const int tid  = threadIdx.x;
  const int wave = tid >> 5;
  const int lane = tid & 31;
  const int lo   = lane & 15;
  const int hi8  = (lane >> 4) << 3;
  const int wm   = wave & 3;   // 0..3 : 32-row M sub-tile
  const int wn   = wave >> 2;  // 0..1 : 32-col N sub-tile
  const int m0   = blockIdx.y * 128;
  const int n0   = blockIdx.x * 64;

  v8f acc[2][2];
  #pragma unroll
  for (int mi = 0; mi < 2; ++mi)
    #pragma unroll
    for (int ni = 0; ni < 2; ++ni)
      acc[mi][ni] = (v8f){0.f,0.f,0.f,0.f,0.f,0.f,0.f,0.f};

  for (int kc = 0; kc < K; kc += KC) {
    // Prefetch next K-chunk into cache while we stage this one.
    if (kc + KC < K) {
      int pr = tid >> 1, ph = (tid & 1) << 5;        // 128 rows x 2 halves
      __builtin_prefetch(A + (size_t)(m0 + pr) * lda + kc + KC + ph, 0, 1);
      if (tid < 128) {
        int wr = tid >> 1, wh = (tid & 1) << 5;      // 64 rows x 2 halves
        __builtin_prefetch(W + (size_t)(n0 + wr) * ldw + kc + KC + wh, 0, 1);
      }
    }
    __syncthreads();
    // Stage A tile (128x64 f32 -> bf16): 8 float4 per thread, coalesced.
    #pragma unroll
    for (int j = 0; j < 8; ++j) {
      int u   = j * 256 + tid;
      int row = u >> 4;
      int kq  = (u & 15) << 2;
      const float4 av = *(const float4*)(A + (size_t)(m0 + row) * lda + kc + kq);
      As[row][kq + 0] = (__bf16)av.x; As[row][kq + 1] = (__bf16)av.y;
      As[row][kq + 2] = (__bf16)av.z; As[row][kq + 3] = (__bf16)av.w;
    }
    // Stage W tile (64x64): 4 float4 per thread.
    #pragma unroll
    for (int j = 0; j < 4; ++j) {
      int u   = j * 256 + tid;
      int row = u >> 4;
      int kq  = (u & 15) << 2;
      const float4 wv = *(const float4*)(W + (size_t)(n0 + row) * ldw + kc + kq);
      Ws[row][kq + 0] = (__bf16)wv.x; Ws[row][kq + 1] = (__bf16)wv.y;
      Ws[row][kq + 2] = (__bf16)wv.z; Ws[row][kq + 3] = (__bf16)wv.w;
    }
    __syncthreads();

    #pragma unroll
    for (int c = 0; c < 2; ++c) {            // two 32-deep WMMA steps
      v16bf a0, a1, b0, b1;
      #pragma unroll
      for (int i = 0; i < 16; ++i) {
        int kk = c * 32 + kpat(i, hi8);
        a0[i] = As[wm * 32 + lo][kk];
        a1[i] = As[wm * 32 + 16 + lo][kk];
        b0[i] = Ws[wn * 32 + lo][kk];
        b1[i] = Ws[wn * 32 + 16 + lo][kk];
      }
      acc[0][0] = wmma_bf16(a0, b0, acc[0][0]);
      acc[0][1] = wmma_bf16(a0, b1, acc[0][1]);
      acc[1][0] = wmma_bf16(a1, b0, acc[1][0]);
      acc[1][1] = wmma_bf16(a1, b1, acc[1][1]);
    }
  }

  // Epilogue. C layout: lane holds col n=lo, rows r+hi8 in elements 0..7.
  #pragma unroll
  for (int mi = 0; mi < 2; ++mi) {
    #pragma unroll
    for (int ni = 0; ni < 2; ++ni) {
      #pragma unroll
      for (int r = 0; r < 8; ++r) {
        int m = m0 + wm * 32 + mi * 16 + hi8 + r;
        int n = n0 + wn * 32 + ni * 16 + lo;
        float v = acc[mi][ni][r];
        if (bias) v += bias[n];
        if (R)    v += R[(size_t)m * N + n];
        if (flags & GF_GELU) v = 0.5f * v * (1.f + erff(v * 0.70710678118654752f));
        size_t off = (size_t)m * N + n;
        if (flags & GF_ACCUM) v += C[off];
        C[off] = v;
      }
    }
  }
}

// ---------------------------------------------------------------------------
// Flash attention: per-wave 16-row q tile, online softmax, DH=64 heads.
// Q/K/V/O are [B, S, D_MODEL] with heads packed as h*64+d.
// grid = (S/16/8, H, B), block = 256 (8 waves).
// ---------------------------------------------------------------------------
__global__ __launch_bounds__(256)
void attn_kernel(const float* __restrict__ Q, const float* __restrict__ Kb,
                 const float* __restrict__ V, float* __restrict__ O)
{
  __shared__ __bf16 P[8][16][32];  // per-wave probability tile (C->A relayout)

  const int tid  = threadIdx.x;
  const int wave = tid >> 5;
  const int lane = tid & 31;
  const int lo   = lane & 15;
  const int hi8  = (lane >> 4) << 3;
  const int q0   = (blockIdx.x * 8 + wave) * 16;
  const int h    = blockIdx.y;
  const int b    = blockIdx.z;
  const size_t base = ((size_t)b * SEQ) * D_MODEL + (size_t)h * DHEAD;

  // Persistent q operand: A layout, 16x64 split into two K=32 chunks.
  v16bf aq0, aq1;
  #pragma unroll
  for (int i = 0; i < 16; ++i) {
    int kk = kpat(i, hi8);
    size_t qrow = base + (size_t)(q0 + lo) * D_MODEL;
    aq0[i] = (__bf16)Q[qrow + kk];
    aq1[i] = (__bf16)Q[qrow + 32 + kk];
  }

  v8f acc[4];
  #pragma unroll
  for (int t = 0; t < 4; ++t) acc[t] = (v8f){0.f,0.f,0.f,0.f,0.f,0.f,0.f,0.f};
  float mrow[8], lrow[8];
  #pragma unroll
  for (int r = 0; r < 8; ++r) { mrow[r] = -1e30f; lrow[r] = 0.f; }

  for (int kt = 0; kt < SEQ; kt += 32) {
    // Scores: two 16x16 tiles (keys kt..kt+15, kt+16..kt+31), K=64 in 2 WMMAs.
    v8f s[2];
    #pragma unroll
    for (int j = 0; j < 2; ++j) {
      v16bf bk0, bk1;
      size_t krow = base + (size_t)(kt + j * 16 + lo) * D_MODEL;
      #pragma unroll
      for (int i = 0; i < 16; ++i) {
        int kk = kpat(i, hi8);
        bk0[i] = (__bf16)Kb[krow + kk];
        bk1[i] = (__bf16)Kb[krow + 32 + kk];
      }
      v8f z = {0.f,0.f,0.f,0.f,0.f,0.f,0.f,0.f};
      z = wmma_bf16(aq0, bk0, z);
      z = wmma_bf16(aq1, bk1, z);
      s[j] = z;
    }

    // Online softmax over the 32 new columns. Each row lives in 16 lanes of
    // one half-wave -> shfl_xor reductions with masks 8..1 stay in-group.
    #pragma unroll
    for (int r = 0; r < 8; ++r) {
      float x0 = s[0][r] * 0.125f;   // 1/sqrt(64)
      float x1 = s[1][r] * 0.125f;
      float mx = fmaxf(x0, x1);
      #pragma unroll
      for (int d = 8; d >= 1; d >>= 1) mx = fmaxf(mx, __shfl_xor(mx, d, 32));
      float mnew  = fmaxf(mrow[r], mx);
      float alpha = __expf(mrow[r] - mnew);
      float p0 = __expf(x0 - mnew);
      float p1 = __expf(x1 - mnew);
      float ps = p0 + p1;
      #pragma unroll
      for (int d = 8; d >= 1; d >>= 1) ps += __shfl_xor(ps, d, 32);
      lrow[r] = lrow[r] * alpha + ps;
      mrow[r] = mnew;
      #pragma unroll
      for (int t = 0; t < 4; ++t) acc[t][r] *= alpha;
      // store P in (row, col) form; same-wave LDS round-trip re-layouts C->A
      P[wave][hi8 + r][lo]      = (__bf16)p0;
      P[wave][hi8 + r][16 + lo] = (__bf16)p1;
    }

    v16bf ap;
    #pragma unroll
    for (int i = 0; i < 16; ++i) ap[i] = P[wave][lo][kpat(i, hi8)];

    // ctx += P[16x32] @ V[32x64] : 4 WMMAs over the 4 N-subtiles of DH.
    #pragma unroll
    for (int t = 0; t < 4; ++t) {
      v16bf bv;
      #pragma unroll
      for (int i = 0; i < 16; ++i) {
        int kk = kpat(i, hi8);
        bv[i] = (__bf16)V[base + (size_t)(kt + kk) * D_MODEL + t * 16 + lo];
      }
      acc[t] = wmma_bf16(ap, bv, acc[t]);
    }
  }

  #pragma unroll
  for (int r = 0; r < 8; ++r) {
    float inv = 1.f / lrow[r];
    size_t orow = base + (size_t)(q0 + hi8 + r) * D_MODEL;
    #pragma unroll
    for (int t = 0; t < 4; ++t) O[orow + t * 16 + lo] = acc[t][r] * inv;
  }
}

// ---------------------------------------------------------------------------
// LayerNorm, torch-custom flavor: std uses Bessel (N-1), eps added to std.
// One block (256 threads) per row of 768.
// ---------------------------------------------------------------------------
__global__ __launch_bounds__(256)
void ln_kernel(const float* __restrict__ X, const float* __restrict__ G,
               const float* __restrict__ Bt, float* __restrict__ Y)
{
  __shared__ float red[256];
  const int row = blockIdx.x;
  const int t   = threadIdx.x;
  const float* x = X + (size_t)row * D_MODEL;

  float a0 = x[t], a1 = x[t + 256], a2 = x[t + 512];
  red[t] = a0 + a1 + a2;
  __syncthreads();
  for (int off = 128; off > 0; off >>= 1) {
    if (t < off) red[t] += red[t + off];
    __syncthreads();
  }
  float mean = red[0] * (1.f / (float)D_MODEL);
  __syncthreads();

  float c0 = a0 - mean, c1 = a1 - mean, c2 = a2 - mean;
  red[t] = c0 * c0 + c1 * c1 + c2 * c2;
  __syncthreads();
  for (int off = 128; off > 0; off >>= 1) {
    if (t < off) red[t] += red[t + off];
    __syncthreads();
  }
  float var   = red[0] * (1.f / (float)(D_MODEL - 1));
  float denom = sqrtf(var) + 1e-6f;
  float inv   = 1.f / denom;

  float* y = Y + (size_t)row * D_MODEL;
  y[t]       = G[t]       * c0 * inv + Bt[t];
  y[t + 256] = G[t + 256] * c1 * inv + Bt[t + 256];
  y[t + 512] = G[t + 512] * c2 * inv + Bt[t + 512];
}

// ---------------------------------------------------------------------------
// Transpose first two dims: X[X0,X1,D] -> Y[X1,X0,D].
// ---------------------------------------------------------------------------
__global__ __launch_bounds__(256)
void transpose01_kernel(const float* __restrict__ X, float* __restrict__ Y,
                        int X0, int X1, int Dd)
{
  size_t idx   = (size_t)blockIdx.x * 256 + threadIdx.x;
  size_t total = (size_t)X0 * X1 * Dd;
  if (idx >= total) return;
  int d = (int)(idx % Dd);
  size_t r = idx / Dd;
  int j = (int)(r % X1);
  int i = (int)(r / X1);
  Y[((size_t)j * X0 + i) * Dd + d] = X[idx];
}

// ---------------------------------------------------------------------------
// Host side
// ---------------------------------------------------------------------------
static inline void launch_gemm(hipStream_t s, const float* A, int lda,
                               const float* W, int ldw, const float* bias,
                               const float* R, float* C,
                               int M, int N, int K, int flags)
{
  dim3 grid(N / 64, M / 128);
  gemm_wmma_kernel<<<grid, 256, 0, s>>>(A, lda, W, ldw, bias, R, C, M, N, K, flags);
}

extern "C" void kernel_launch(void* const* d_in, const int* in_sizes, int n_in,
                              void* d_out, int out_size, void* d_ws, size_t ws_size,
                              hipStream_t stream)
{
  (void)in_sizes; (void)n_in; (void)out_size; (void)ws_size;

  const float* e    = (const float*)d_in[0];
  const float* f    = (const float*)d_in[1];
  const float* wq   = (const float*)d_in[2];
  const float* wk   = (const float*)d_in[3];
  const float* wv   = (const float*)d_in[4];
  const float* wo   = (const float*)d_in[5];
  const float* bq   = (const float*)d_in[6];
  const float* bk   = (const float*)d_in[7];
  const float* bvp  = (const float*)d_in[8];
  const float* bo   = (const float*)d_in[9];
  const float* alg  = (const float*)d_in[10];
  const float* alb  = (const float*)d_in[11];
  const float* fw1  = (const float*)d_in[12];
  const float* fb1  = (const float*)d_in[13];
  const float* fw2  = (const float*)d_in[14];
  const float* fb2  = (const float*)d_in[15];
  const float* flg  = (const float*)d_in[16];
  const float* flb  = (const float*)d_in[17];
  const float* pw   = (const float*)d_in[18];
  const float* pb   = (const float*)d_in[19];
  const float* finw = (const float*)d_in[20];
  const float* finb = (const float*)d_in[21];
  float* out = (float*)d_out;

  const int    M   = BATCH * SEQ;               // 2048
  const size_t NEL = (size_t)M * D_MODEL;       // 1,572,864 floats per buffer
  const int    DD  = D_MODEL * D_MODEL;
  const int    nTblk = (int)((NEL + 255) / 256);

  float* buf[10];
  for (int i = 0; i < 10; ++i) buf[i] = (float*)d_ws + (size_t)i * NEL;

  for (int ti = 0; ti < 2; ++ti) {
    const float* src  = (ti == 0) ? e : f;
    const float* src1 = (ti == 0) ? f : e;
    const int l0 = ti * 2 + 0;
    const int l1 = ti * 2 + 1;

    // ---- self-attention encoder block (layer 0) ----
    transpose01_kernel<<<nTblk, 256, 0, stream>>>(src, buf[0], SEQ, BATCH, D_MODEL);
    launch_gemm(stream, buf[0], D_MODEL, wq + (size_t)l0 * DD, D_MODEL,
                bq + l0 * D_MODEL, nullptr, buf[1], M, D_MODEL, D_MODEL, 0);
    launch_gemm(stream, buf[0], D_MODEL, wk + (size_t)l0 * DD, D_MODEL,
                bk + l0 * D_MODEL, nullptr, buf[2], M, D_MODEL, D_MODEL, 0);
    launch_gemm(stream, buf[0], D_MODEL, wv + (size_t)l0 * DD, D_MODEL,
                bvp + l0 * D_MODEL, nullptr, buf[3], M, D_MODEL, D_MODEL, 0);
    attn_kernel<<<dim3(SEQ / 16 / 8, NHEAD, BATCH), 256, 0, stream>>>(buf[1], buf[2], buf[3], buf[4]);
    launch_gemm(stream, buf[4], D_MODEL, wo + (size_t)l0 * DD, D_MODEL,
                bo + l0 * D_MODEL, buf[0], buf[5], M, D_MODEL, D_MODEL, 0);
    ln_kernel<<<M, 256, 0, stream>>>(buf[5], alg + l0 * D_MODEL, alb + l0 * D_MODEL, buf[6]);
    launch_gemm(stream, buf[6], D_MODEL, fw1 + (size_t)l0 * DD, D_MODEL,
                fb1 + l0 * D_MODEL, nullptr, buf[7], M, D_MODEL, D_MODEL, GF_GELU);
    launch_gemm(stream, buf[7], D_MODEL, fw2 + (size_t)l0 * DD, D_MODEL,
                fb2 + l0 * D_MODEL, buf[6], buf[5], M, D_MODEL, D_MODEL, 0);
    ln_kernel<<<M, 256, 0, stream>>>(buf[5], flg + l0 * D_MODEL, flb + l0 * D_MODEL, buf[8]); // feats

    // ---- cross-attention encoder block (layer 1) ----
    transpose01_kernel<<<nTblk, 256, 0, stream>>>(src1, buf[0], SEQ, BATCH, D_MODEL);
    launch_gemm(stream, buf[0], D_MODEL, wq + (size_t)l1 * DD, D_MODEL,
                bq + l1 * D_MODEL, nullptr, buf[1], M, D_MODEL, D_MODEL, 0);
    launch_gemm(stream, buf[8], D_MODEL, wk + (size_t)l1 * DD, D_MODEL,
                bk + l1 * D_MODEL, nullptr, buf[2], M, D_MODEL, D_MODEL, 0);
    launch_gemm(stream, buf[8], D_MODEL, wv + (size_t)l1 * DD, D_MODEL,
                bvp + l1 * D_MODEL, nullptr, buf[3], M, D_MODEL, D_MODEL, 0);
    attn_kernel<<<dim3(SEQ / 16 / 8, NHEAD, BATCH), 256, 0, stream>>>(buf[1], buf[2], buf[3], buf[4]);
    launch_gemm(stream, buf[4], D_MODEL, wo + (size_t)l1 * DD, D_MODEL,
                bo + l1 * D_MODEL, buf[0], buf[5], M, D_MODEL, D_MODEL, 0);
    ln_kernel<<<M, 256, 0, stream>>>(buf[5], alg + l1 * D_MODEL, alb + l1 * D_MODEL, buf[6]);
    launch_gemm(stream, buf[6], D_MODEL, fw1 + (size_t)l1 * DD, D_MODEL,
                fb1 + l1 * D_MODEL, nullptr, buf[7], M, D_MODEL, D_MODEL, GF_GELU);
    launch_gemm(stream, buf[7], D_MODEL, fw2 + (size_t)l1 * DD, D_MODEL,
                fb2 + l1 * D_MODEL, buf[6], buf[5], M, D_MODEL, D_MODEL, 0);
    ln_kernel<<<M, 256, 0, stream>>>(buf[5], flg + l1 * D_MODEL, flb + l1 * D_MODEL, buf[9]); // ctx

    // ---- pooling: flat ctx viewed as [B*D, S] rows, @ pool_w.T + pool_b ----
    launch_gemm(stream, buf[9], SEQ, pw + (size_t)ti * SEQ * SEQ, SEQ,
                pb + ti * SEQ, nullptr, buf[4],
                (int)(NEL / SEQ), SEQ, SEQ, 0);   // M=1536, N=K=1024

    // ---- final: [feats | app] @ finw.T + finb, split over K ----
    launch_gemm(stream, buf[8], D_MODEL, finw + (size_t)ti * D_MODEL * 2 * D_MODEL,
                2 * D_MODEL, finb + ti * D_MODEL, nullptr, buf[5],
                M, D_MODEL, D_MODEL, 0);
    launch_gemm(stream, buf[4], D_MODEL,
                finw + (size_t)ti * D_MODEL * 2 * D_MODEL + D_MODEL,
                2 * D_MODEL, nullptr, nullptr, buf[5],
                M, D_MODEL, D_MODEL, GF_ACCUM);

    // ---- [B,S,D] -> [S,B,D] into the output slot ----
    transpose01_kernel<<<nTblk, 256, 0, stream>>>(buf[5], out + (size_t)ti * NEL,
                                                  BATCH, SEQ, D_MODEL);
  }
}
